// MultiHeadSelfAttention_87393994539169
// MI455X (gfx1250) — compile-verified, matrix operand order checked by reading, not run on hardware
//
#include <hip/hip_runtime.h>
#include <hip/hip_bf16.h>
#include <math.h>

// ---------------------------------------------------------------------------
// MI455X (gfx1250) causal multi-head attention forward.
//   x:[4,2048,2048] f32, wq/wk/wv/wo:[2048,2048] f32 (out = A @ W^T), 16 heads
// Compute-bound (~412 GFLOP vs ~10us of HBM traffic) -> bf16 WMMA + f32 accum.
// Pipeline:
//   cvt      : f32 -> bf16 one-shot (x, 4 weights)           (~4us bandwidth)
//   gemm<0/1>: Q/K proj + RoPE -> bf16 [bh][s][dk]           (async-LDS staged)
//   gemm<2>  : V proj -> bf16 transposed [bh][dk][s]
//   attn     : wave-level flash attention -> ctx bf16
//   gemm<3>  : output proj -> f32 d_out
// Workspace (~202 MB): xb | wb[4] | Qb | Kb | Vt | ctxb   (all bf16)
// ---------------------------------------------------------------------------

typedef __bf16 bf16;
typedef __attribute__((ext_vector_type(16))) __bf16 bf16x16;
typedef __attribute__((ext_vector_type(8)))  __bf16 bf16x8;
typedef __attribute__((ext_vector_type(4)))  __bf16 bf16x4;
typedef __attribute__((ext_vector_type(8)))  float   f32x8;
typedef __attribute__((ext_vector_type(4)))  float   f32x4;

#define B_   4
#define S_   2048
#define DM   2048
#define H_   16
#define DK   128
#define M_   (B_*S_)   // 8192 rows

static __device__ __forceinline__ bf16x16 frag_cat(bf16x8 lo, bf16x8 hi) {
  return __builtin_shufflevector(lo, hi, 0,1,2,3,4,5,6,7,8,9,10,11,12,13,14,15);
}
static __device__ __forceinline__ f32x8 wmma_bf16(bf16x16 a, bf16x16 b, f32x8 c) {
  // D = A(16x32) * B(32x16) + C, f32 accumulate
  return __builtin_amdgcn_wmma_f32_16x16x32_bf16(false, a, false, b, (short)0, c,
                                                 false, false);
}

// Raw 32-bit LDS offset of a __shared__ object (hardware adds LDS_BASE).
typedef const void __attribute__((address_space(3))) lds_cv;
static __device__ __forceinline__ unsigned lds_off(const void* p) {
  return (unsigned)(size_t)(lds_cv*)p;
}
// Async Global -> LDS, 16B per lane, GVS addressing (SGPR base + VGPR byte off).
// Tracked by ASYNCcnt (CDNA5 Tensor/Async path), not LOADcnt.
static __device__ __forceinline__ void async_b128(unsigned lds, unsigned goff,
                                                  const bf16* base) {
  asm volatile("global_load_async_to_lds_b128 %0, %1, %2"
               :: "v"(lds), "v"(goff), "s"(base) : "memory");
}
static __device__ __forceinline__ void wait_async0() {
  asm volatile("s_wait_asynccnt 0x0" ::: "memory");
}

// ---------------------------------------------------------------------------
// One-shot f32 -> bf16 conversion (vector of 4 per thread).
// ---------------------------------------------------------------------------
__global__ __launch_bounds__(256)
void cvt_f32_bf16(const float* __restrict__ in, bf16* __restrict__ out, int n4) {
  int i = blockIdx.x * 256 + threadIdx.x;
  if (i < n4) {
    f32x4 v = *(const f32x4*)(in + 4 * (size_t)i);
    bf16x4 h;
    h[0] = (bf16)v[0]; h[1] = (bf16)v[1]; h[2] = (bf16)v[2]; h[3] = (bf16)v[3];
    *(bf16x4*)(out + 4 * (size_t)i) = h;
  }
}

// ---------------------------------------------------------------------------
// C = A[M x K] * W[N x K]^T (bf16 in, K = 2048). Block 128x64, 8 waves,
// BK = 64, double-buffered LDS filled by async global->LDS b128 copies.
// MODE 0: Q -> RoPE -> bf16 [bh][s][dk]
// MODE 1: K -> RoPE -> bf16 [bh][s][dk]
// MODE 2: V -> bf16 transposed [bh][dk][s]
// MODE 3: O -> f32 [m][e] (d_out)
// ---------------------------------------------------------------------------
template<int MODE>
__global__ __launch_bounds__(256)
void gemm_xwt(const bf16* __restrict__ A, const bf16* __restrict__ W,
              bf16* __restrict__ outb, float* __restrict__ outf) {
  constexpr int BM = 128, BN = 64, BK = 64, LDP = 72; // 144B pitch (16B mult)
  __shared__ __align__(16) bf16 sA[2][BM][LDP];
  __shared__ __align__(16) bf16 sB[2][BN][LDP];

  const int tid  = threadIdx.x;
  const int lane = tid & 31;
  const int wave = tid >> 5;
  const int ln   = lane & 15;
  const int hi   = lane >> 4;
  const int wm   = (wave & 3) * 32;   // 4 waves down M
  const int wn   = (wave >> 2) * 32;  // 2 waves across N
  const int m0   = blockIdx.x * BM;
  const int n0   = blockIdx.y * BN;

  // stage tile k0 into buffer p: per-lane async b128 copies, no VGPR data
  auto stage = [&](int p, int k0) {
    #pragma unroll
    for (int j = 0; j < 4; ++j) {               // A: 128x64 = 1024 x 16B
      int seg = tid + 256 * j;
      int r = seg >> 3, cs = (seg & 7) * 8;
      async_b128(lds_off(&sA[p][r][cs]),
                 (unsigned)((((m0 + r) * DM) + k0 + cs) * 2), A);
    }
    #pragma unroll
    for (int j = 0; j < 2; ++j) {               // W: 64x64 = 512 x 16B
      int seg = tid + 256 * j;
      int r = seg >> 3, cs = (seg & 7) * 8;
      async_b128(lds_off(&sB[p][r][cs]),
                 (unsigned)((((n0 + r) * DM) + k0 + cs) * 2), W);
    }
  };

  f32x8 c00 = {}, c01 = {}, c10 = {}, c11 = {};

  stage(0, 0);
  wait_async0();
  __syncthreads();

  int p = 0;
  for (int k0 = 0; k0 < DM; k0 += BK) {
    if (k0 + BK < DM) stage(p ^ 1, k0 + BK);    // prefetch next tile (async)
    #pragma unroll
    for (int kc = 0; kc < 2; ++kc) {            // two 32-deep chunks
      const int cb = kc * 32;
      bf16x16 a0 = frag_cat(*(const bf16x8*)&sA[p][wm + ln][cb + hi * 8],
                            *(const bf16x8*)&sA[p][wm + ln][cb + 16 + hi * 8]);
      bf16x16 a1 = frag_cat(*(const bf16x8*)&sA[p][wm + 16 + ln][cb + hi * 8],
                            *(const bf16x8*)&sA[p][wm + 16 + ln][cb + 16 + hi * 8]);
      bf16x16 b0 = frag_cat(*(const bf16x8*)&sB[p][wn + ln][cb + hi * 8],
                            *(const bf16x8*)&sB[p][wn + ln][cb + 16 + hi * 8]);
      bf16x16 b1 = frag_cat(*(const bf16x8*)&sB[p][wn + 16 + ln][cb + hi * 8],
                            *(const bf16x8*)&sB[p][wn + 16 + ln][cb + 16 + hi * 8]);
      c00 = wmma_bf16(a0, b0, c00);
      c01 = wmma_bf16(a0, b1, c01);
      c10 = wmma_bf16(a1, b0, c10);
      c11 = wmma_bf16(a1, b1, c11);
    }
    wait_async0();          // own async copies for next tile have landed
    __syncthreads();        // all waves done reading buf p / writing buf p^1
    p ^= 1;
  }

  // ---- epilogue ----
  #pragma unroll
  for (int mi = 0; mi < 2; ++mi) {
    #pragma unroll
    for (int ni = 0; ni < 2; ++ni) {
      f32x8 cv = (mi == 0) ? ((ni == 0) ? c00 : c01) : ((ni == 0) ? c10 : c11);
      const int eg    = n0 + wn + ni * 16 + ln;       // output feature
      const int mrow0 = m0 + wm + mi * 16 + 8 * hi;   // row of v=0
      if (MODE == 3) {
        #pragma unroll
        for (int v = 0; v < 8; ++v)
          outf[(size_t)(mrow0 + v) * DM + eg] = cv[v];
      } else if (MODE == 2) {
        const int bh = ((mrow0 >> 11) << 4) + (eg >> 7);
        const int sb = mrow0 & (S_ - 1);
        const int d  = eg & (DK - 1);
        bf16x8 pk;
        #pragma unroll
        for (int v = 0; v < 8; ++v) pk[v] = (bf16)cv[v];
        *(bf16x8*)(outb + ((size_t)bh * DK + d) * S_ + sb) = pk;  // transposed
      } else {  // Q or K with interleaved RoPE
        const int bh = ((mrow0 >> 11) << 4) + (eg >> 7);
        const int sb = mrow0 & (S_ - 1);
        const int d  = eg & (DK - 1);
        const int i2 = d >> 1;
        // inv_freq = 10000^(-2*i2/128) = exp(-i2 * ln(1e4)/64)
        const float inv = __expf(-0.14391156f * (float)i2);
        #pragma unroll
        for (int v = 0; v < 8; ++v) {
          float val = cv[v];
          float par = __shfl_xor(val, 1, 32);  // pair lane (d ^ 1)
          float ang = (float)(sb + v) * inv;
          float sn, cs;
          __sincosf(ang, &sn, &cs);
          float r = (d & 1) ? fmaf(par, sn, val * cs)    // odd:  x1*sin + x2*cos
                            : fmaf(val, cs, -par * sn);  // even: x1*cos - x2*sin
          outb[((size_t)bh * S_ + (sb + v)) * DK + d] = (bf16)r;
        }
      }
    }
  }
}

// ---------------------------------------------------------------------------
// Flash attention, causal. Each wave owns 16 query rows; 8 waves/WG = 128 rows.
// Per 32-key chunk: 8 WMMA for S = Q*K^T (dk=128), online softmax with 16-lane
// shfl reductions, P transposed through wave-private LDS, 8 WMMA for PV.
// K/V are L2-resident (1 MB per head, 192 MB L2) -> direct clause-grouped loads.
// ---------------------------------------------------------------------------
__global__ __launch_bounds__(256)
void attn_fwd(const bf16* __restrict__ Qb, const bf16* __restrict__ Kb,
              const bf16* __restrict__ Vt, bf16* __restrict__ ctxb) {
  __shared__ __align__(16) bf16 sP[8][16][48];  // per-wave P tile, 96B pitch

  const int tid = threadIdx.x, lane = tid & 31, w = tid >> 5;
  const int ln = lane & 15, hi = lane >> 4;
  const int bh = blockIdx.y;
  const int b  = bh >> 4, h = bh & 15;
  const int q0 = blockIdx.x * 128 + w * 16;
  const float scale = 0.08838834764831845f;  // 1/sqrt(128)

  // Q fragments resident for the whole key loop (4 chunks of 32 over dk=128)
  const bf16* qrow = Qb + ((size_t)bh * S_ + q0 + ln) * DK;
  bf16x16 aq[4];
  #pragma unroll
  for (int kc = 0; kc < 4; ++kc)
    aq[kc] = frag_cat(*(const bf16x8*)(qrow + kc * 32 + hi * 8),
                      *(const bf16x8*)(qrow + kc * 32 + 16 + hi * 8));

  f32x8 o[8];
  float mi[8], li[8];
  {
    f32x8 z = {};
    #pragma unroll
    for (int j = 0; j < 8; ++j) o[j] = z;
  }
  #pragma unroll
  for (int v = 0; v < 8; ++v) { mi[v] = -3.0e38f; li[v] = 0.f; }

  const int qlast = q0 + 15;
  for (int kb = 0; kb <= qlast; kb += 32) {
    // ---- S = Q * K^T for 32 keys ----
    f32x8 s0 = {}, s1 = {};
    const bf16* k0p = Kb + ((size_t)bh * S_ + kb + ln) * DK;
    const bf16* k1p = k0p + 16 * DK;
    #pragma unroll
    for (int kc = 0; kc < 4; ++kc) {
      bf16x16 bk0 = frag_cat(*(const bf16x8*)(k0p + kc * 32 + hi * 8),
                             *(const bf16x8*)(k0p + kc * 32 + 16 + hi * 8));
      s0 = wmma_bf16(aq[kc], bk0, s0);
      bf16x16 bk1 = frag_cat(*(const bf16x8*)(k1p + kc * 32 + hi * 8),
                             *(const bf16x8*)(k1p + kc * 32 + 16 + hi * 8));
      s1 = wmma_bf16(aq[kc], bk1, s1);
    }
    __builtin_prefetch(Kb + ((size_t)bh * S_ + kb + 32 + ln) * DK, 0, 1);

    // ---- online softmax (rows live in 16-lane halves, 8 rows/lane) ----
    #pragma unroll
    for (int v = 0; v < 8; ++v) {
      const int qg = q0 + v + 8 * hi;
      float x0 = (kb + ln      <= qg) ? s0[v] * scale : -1.0e30f;
      float x1 = (kb + 16 + ln <= qg) ? s1[v] * scale : -1.0e30f;
      float rm = fmaxf(x0, x1);
      rm = fmaxf(rm, __shfl_xor(rm, 1, 32));
      rm = fmaxf(rm, __shfl_xor(rm, 2, 32));
      rm = fmaxf(rm, __shfl_xor(rm, 4, 32));
      rm = fmaxf(rm, __shfl_xor(rm, 8, 32));
      float mnew  = fmaxf(mi[v], rm);
      float alpha = __expf(mi[v] - mnew);
      float p0 = __expf(x0 - mnew);
      float p1 = __expf(x1 - mnew);
      float rs = p0 + p1;
      rs += __shfl_xor(rs, 1, 32);
      rs += __shfl_xor(rs, 2, 32);
      rs += __shfl_xor(rs, 4, 32);
      rs += __shfl_xor(rs, 8, 32);
      li[v] = li[v] * alpha + rs;
      mi[v] = mnew;
      #pragma unroll
      for (int j = 0; j < 8; ++j) o[j][v] *= alpha;
      const int m = v + 8 * hi;
      sP[w][m][ln]      = (bf16)p0;   // C-layout -> LDS (transpose for A-frag)
      sP[w][m][16 + ln] = (bf16)p1;
    }

    // ---- PV: A = P(16x32) from LDS, B = Vt chunk (two 16B loads per frag) ----
    bf16x16 pf = frag_cat(*(const bf16x8*)&sP[w][ln][hi * 8],
                          *(const bf16x8*)&sP[w][ln][16 + hi * 8]);
    const bf16* vp = Vt + ((size_t)bh * DK + ln) * S_ + kb;
    #pragma unroll
    for (int j = 0; j < 8; ++j) {
      const bf16* vq = vp + (size_t)j * 16 * S_;
      bf16x16 bv = frag_cat(*(const bf16x8*)(vq + hi * 8),
                            *(const bf16x8*)(vq + 16 + hi * 8));
      o[j] = wmma_bf16(pf, bv, o[j]);
    }
  }

  // ---- normalize + write ctx bf16 [b][s][h*128+d] ----
  #pragma unroll
  for (int v = 0; v < 8; ++v) {
    const float invl = 1.0f / li[v];
    const int qg = q0 + v + 8 * hi;
    bf16* op = ctxb + ((size_t)b * S_ + qg) * DM + h * DK + ln;
    #pragma unroll
    for (int j = 0; j < 8; ++j) op[j * 16] = (bf16)(o[j][v] * invl);
  }
}

// ---------------------------------------------------------------------------
extern "C" void kernel_launch(void* const* d_in, const int* in_sizes, int n_in,
                              void* d_out, int out_size, void* d_ws, size_t ws_size,
                              hipStream_t stream) {
  const float* x  = (const float*)d_in[0];
  const float* wq = (const float*)d_in[1];
  const float* wk = (const float*)d_in[2];
  const float* wv = (const float*)d_in[3];
  const float* wo = (const float*)d_in[4];
  float* out = (float*)d_out;

  const size_t xe = (size_t)M_ * DM;       // 16,777,216 elems
  const size_t we = (size_t)DM * DM;       //  4,194,304 elems
  bf16* xb   = (bf16*)d_ws;
  bf16* wqb  = xb  + xe;
  bf16* wkb  = wqb + we;
  bf16* wvb  = wkb + we;
  bf16* wob  = wvb + we;
  bf16* Qb   = wob + we;
  bf16* Kb   = Qb  + xe;
  bf16* Vt   = Kb  + xe;
  bf16* ctxb = Vt  + xe;

  cvt_f32_bf16<<<(int)(xe / 4 / 256), 256, 0, stream>>>(x,  xb,  (int)(xe / 4));
  cvt_f32_bf16<<<(int)(we / 4 / 256), 256, 0, stream>>>(wq, wqb, (int)(we / 4));
  cvt_f32_bf16<<<(int)(we / 4 / 256), 256, 0, stream>>>(wk, wkb, (int)(we / 4));
  cvt_f32_bf16<<<(int)(we / 4 / 256), 256, 0, stream>>>(wv, wvb, (int)(we / 4));
  cvt_f32_bf16<<<(int)(we / 4 / 256), 256, 0, stream>>>(wo, wob, (int)(we / 4));

  dim3 gg(M_ / 128, DM / 64), gb(256);
  gemm_xwt<0><<<gg, gb, 0, stream>>>(xb, wqb, Qb, nullptr);
  gemm_xwt<1><<<gg, gb, 0, stream>>>(xb, wkb, Kb, nullptr);
  gemm_xwt<2><<<gg, gb, 0, stream>>>(xb, wvb, Vt, nullptr);
  attn_fwd<<<dim3(S_ / 128, B_ * H_), 256, 0, stream>>>(Qb, Kb, Vt, ctxb);
  gemm_xwt<3><<<gg, gb, 0, stream>>>(ctxb, wob, nullptr, out);
}
